// SubgraphEncoder_2216203125208
// MI455X (gfx1250) — matrix operand order repaired and checked by reading, not compiled, not run on hardware
//
#include <hip/hip_runtime.h>
#include <hip/hip_bf16.h>

// SubgraphEncoder forward for MI455X (gfx1250, wave32).
//
// GEMMs: v_wmma_f32_16x16x32_bf16, one wave per 16-row strip x full Nc so
// each A fragment is loaded once and reused across all N tiles (32 unrolled
// WMMAs per wave for Nc=128). Self-loop term of GCN aggregation fused into
// the GEMM epilogue. Aggregation / BN / softmax in f32.
// Exact simplifications: conv1_b/conv2_b/fc_b/bot_b are column constants
// before batch-stat BatchNorm -> cancel; att_b2 cancels in softmax.

#define NN   40000
#define HH   128
#define SUBG 400
#define EPSBN 1e-5f

typedef __bf16 bf16_t;
typedef bf16_t v16bf __attribute__((ext_vector_type(16)));
typedef bf16_t v8bf  __attribute__((ext_vector_type(8)));
typedef float  v8f   __attribute__((ext_vector_type(8)));

static __device__ __forceinline__ bf16_t f2bf(float f) {
    union { float f; unsigned int u; } in; in.f = f;
    unsigned int r = in.u + 0x7FFFu + ((in.u >> 16) & 1u);   // round-nearest-even
    union { unsigned short s; bf16_t b; } out; out.s = (unsigned short)(r >> 16);
    return out.b;
}

static __device__ __forceinline__ void atomicAddF(float* p, float v) {
    __hip_atomic_fetch_add(p, v, __ATOMIC_RELAXED, __HIP_MEMORY_SCOPE_AGENT);
}

// ---------------- small utility kernels ----------------

__global__ void k_zero(float* __restrict__ p, int n) {
    int i = blockIdx.x * blockDim.x + threadIdx.x;
    if (i < n) p[i] = 0.0f;
}

__global__ void k_fill1(float* __restrict__ p, int n) {
    int i = blockIdx.x * blockDim.x + threadIdx.x;
    if (i < n) p[i] = 1.0f;
}

__global__ void k_deg_edges(const int* __restrict__ dst, float* __restrict__ deg, int e) {
    int i = blockIdx.x * blockDim.x + threadIdx.x;
    if (i < e) atomicAddF(&deg[dst[i]], 1.0f);
}

__global__ void k_rsqrt_inplace(float* __restrict__ d, int n) {
    int i = blockIdx.x * blockDim.x + threadIdx.x;
    if (i < n) d[i] = rsqrtf(d[i]);   // deg >= 1 always (self loop)
}

__global__ void k_f32_to_bf16(const float* __restrict__ in, bf16_t* __restrict__ out, int n) {
    int i = blockIdx.x * blockDim.x + threadIdx.x;
    if (i < n) out[i] = f2bf(in[i]);
}

// ---------------- weight packing into WMMA B-fragment layout ----------------
// Fragment (kb, nb): lane l holds column n = nb*16 + (l&15);
// vector element e holds k = kb*32 + (l>>4)*8 + e (e<8), kb*32 + 16 + (l>>4)*8 + (e-8) (e>=8).
__global__ void k_pack_w(const float* __restrict__ W, bf16_t* __restrict__ Bp,
                         int K, int Nc) {
    int idx = blockIdx.x * blockDim.x + threadIdx.x;
    if (idx >= K * Nc) return;
    int e      = idx & 15;
    int lane   = (idx >> 4) & 31;
    int fragid = idx >> 9;
    int NB = Nc >> 4;
    int nb = fragid % NB;
    int kb = fragid / NB;
    int n = nb * 16 + (lane & 15);
    int k = kb * 32 + (lane >> 4) * 8 + (e & 7) + ((e >> 3) << 4);
    Bp[idx] = f2bf(W[k * Nc + n]);
}

// ---------------- WMMA GEMM (strip version) ----------------
// C[M, NT*16] = A[M, KB*32](bf16, row-major) * Bpacked.
// One wave per 16-row strip: A fragment loaded once per K-step, reused
// across all NT N-tiles. Optional fused GCN self-loop epilogue:
//   selfout[row, :] = C[row, :] * dis[row]^2
template <int NT, int KB>
__global__ void __launch_bounds__(256)
k_wmma_gemm_strip(const bf16_t* __restrict__ A, const bf16_t* __restrict__ Bp,
                  float* __restrict__ Cout, int M,
                  const float* __restrict__ dis, float* __restrict__ selfout) {
    constexpr int K  = KB * 32;
    constexpr int Nc = NT * 16;
    const int lane = threadIdx.x & 31;
    const int wave = threadIdx.x >> 5;
    const int mt   = blockIdx.x * 8 + wave;
    const int Mtiles = (M + 15) >> 4;
    if (mt >= Mtiles) return;                 // wave-uniform exit

    const int half = lane >> 4;
    const int l15  = lane & 15;
    const int rowA = mt * 16 + l15;
    const bool rowOK = rowA < M;

    v8f acc[NT];
#pragma unroll
    for (int i = 0; i < NT; ++i) acc[i] = {};

#pragma unroll
    for (int kb = 0; kb < KB; ++kb) {
        // 16-bit A 16x32 layout: elems 0..7  = K[kb*32 + half*8 ..],
        //                        elems 8..15 = K[kb*32 + 16 + half*8 ..]
        v16bf a = {};
        if (rowOK) {
            const bf16_t* ap = A + (size_t)rowA * K + kb * 32 + half * 8;
            v8bf lo = *(const v8bf*)(ap);
            v8bf hi = *(const v8bf*)(ap + 16);
#pragma unroll
            for (int i = 0; i < 8; ++i) { a[i] = lo[i]; a[i + 8] = hi[i]; }
        }
#pragma unroll
        for (int nt = 0; nt < NT; ++nt) {
            v16bf b = *(const v16bf*)(Bp + ((size_t)(kb * NT + nt) * 32 + lane) * 16);
            acc[nt] = __builtin_amdgcn_wmma_f32_16x16x32_bf16(
                          false, a, false, b, (short)0, acc[nt], false, false);
        }
    }

    // C/D layout: VGPR j -> row mt*16 + j + 8*half, col (lane&15) + nt*16
#pragma unroll
    for (int j = 0; j < 8; ++j) {
        int row = mt * 16 + j + half * 8;
        if (row < M) {
            float d2 = 0.0f;
            if (dis) { float d = dis[row]; d2 = d * d; }
#pragma unroll
            for (int nt = 0; nt < NT; ++nt) {
                float v = acc[nt][j];
                size_t o = (size_t)row * Nc + nt * 16 + l15;
                Cout[o] = v;
                if (selfout) selfout[o] = v * d2;
            }
        }
    }
}

// ---------------- GCN edge aggregation (scatter-add) ----------------

__global__ void k_edge_agg(const float* __restrict__ h, const float* __restrict__ dis,
                           const int* __restrict__ src, const int* __restrict__ dst,
                           float* __restrict__ out, int total) {
    int i = blockIdx.x * blockDim.x + threadIdx.x;
    if (i >= total) return;
    int e = i >> 7;
    int c = i & 127;
    int s = src[e], d = dst[e];
    float v = h[(size_t)s * HH + c] * dis[s] * dis[d];
    atomicAddF(&out[(size_t)d * HH + c], v);
}

// ---------------- BatchNorm (batch statistics, biased variance) ----------------

__global__ void k_bn_stats(const float* __restrict__ x, float* __restrict__ sum,
                           float* __restrict__ sumsq, int rows, int cols, int chunk) {
    int c = threadIdx.x;                 // blockDim.x == cols
    int r0 = blockIdx.x * chunk;
    int r1 = r0 + chunk; if (r1 > rows) r1 = rows;
    float s = 0.f, s2 = 0.f;
    for (int r = r0; r < r1; ++r) {
        float v = x[(size_t)r * cols + c];
        s += v; s2 += v * v;
    }
    atomicAddF(sum + c, s);
    atomicAddF(sumsq + c, s2);
}

__global__ void k_bn_final(const float* __restrict__ sum, const float* __restrict__ sumsq,
                           const float* __restrict__ g, const float* __restrict__ b,
                           float* __restrict__ scale, float* __restrict__ shift,
                           int rows, int cols) {
    int c = threadIdx.x;
    if (c >= cols) return;
    float inv = 1.0f / (float)rows;
    float mean = sum[c] * inv;
    float var  = sumsq[c] * inv - mean * mean;
    float sc = g[c] * rsqrtf(var + EPSBN);
    scale[c] = sc;
    shift[c] = b[c] - mean * sc;
}

// BN + ReLU; optional f32 and/or bf16 outputs
__global__ void k_bn_apply_relu(const float* __restrict__ x,
                                const float* __restrict__ scale, const float* __restrict__ shift,
                                float* __restrict__ outf, bf16_t* __restrict__ outbf,
                                int rows, int cols) {
    int i = blockIdx.x * blockDim.x + threadIdx.x;
    if (i >= rows * cols) return;
    int c = i % cols;
    float v = fmaxf(x[i] * scale[c] + shift[c], 0.0f);
    if (outf)  outf[i]  = v;
    if (outbf) outbf[i] = f2bf(v);
}

// ---------------- attention scores: s[n] = sum_j tanh(t[n,j] + b1[j]) * w2[j] ----------------
// (att_b2 cancels inside softmax)
__global__ void k_scores(const float* __restrict__ t, const float* __restrict__ b1,
                         const float* __restrict__ w2, float* __restrict__ scores, int n) {
    int gid  = blockIdx.x * blockDim.x + threadIdx.x;
    int wid  = gid >> 5;
    int lane = gid & 31;
    if (wid >= n) return;
    float acc = 0.f;
    for (int j = lane; j < HH / 2; j += 32)
        acc += tanhf(t[(size_t)wid * (HH / 2) + j] + b1[j]) * w2[j];
#pragma unroll
    for (int off = 16; off > 0; off >>= 1)
        acc += __shfl_xor(acc, off, 32);
    if (lane == 0) scores[wid] = acc;
}

// ---------------- softmax-weighted pooling per cell ----------------
__global__ void __launch_bounds__(128)
k_pool(const float* __restrict__ emb, const float* __restrict__ scores,
       bf16_t* __restrict__ pooledbf) {
    __shared__ float w[SUBG];
    __shared__ float red[128];
    int cell = blockIdx.x, t = threadIdx.x;
    int base = cell * SUBG;

    float m = -3.4e38f;
    for (int s = t; s < SUBG; s += 128) m = fmaxf(m, scores[base + s]);
    red[t] = m; __syncthreads();
    for (int off = 64; off > 0; off >>= 1) {
        if (t < off) red[t] = fmaxf(red[t], red[t + off]);
        __syncthreads();
    }
    float mx = red[0]; __syncthreads();

    float z = 0.f;
    for (int s = t; s < SUBG; s += 128) {
        float e = expf(scores[base + s] - mx);
        w[s] = e; z += e;
    }
    red[t] = z; __syncthreads();
    for (int off = 64; off > 0; off >>= 1) {
        if (t < off) red[t] += red[t + off];
        __syncthreads();
    }
    float invZ = 1.0f / red[0];

    float acc = 0.f;
    for (int s = 0; s < SUBG; ++s)
        acc += w[s] * emb[(size_t)(base + s) * HH + t];
    pooledbf[cell * HH + t] = f2bf(acc * invZ);
}

// ---------------- host orchestration ----------------

#define CDIV(a, b) (((a) + (b) - 1) / (b))

extern "C" void kernel_launch(void* const* d_in, const int* in_sizes, int n_in,
                              void* d_out, int out_size, void* d_ws, size_t ws_size,
                              hipStream_t stream) {
    const float* x       = (const float*)d_in[0];
    const int*   ei      = (const int*)d_in[1];
    const float* conv1_w = (const float*)d_in[2];
    const float* bn1_g   = (const float*)d_in[4];
    const float* bn1_b   = (const float*)d_in[5];
    const float* conv2_w = (const float*)d_in[6];
    const float* bn2_g   = (const float*)d_in[8];
    const float* bn2_b   = (const float*)d_in[9];
    const float* att_w1  = (const float*)d_in[10];
    const float* att_b1  = (const float*)d_in[11];
    const float* att_w2  = (const float*)d_in[12];
    const float* fc_w    = (const float*)d_in[14];
    const float* bne_g   = (const float*)d_in[16];
    const float* bne_b   = (const float*)d_in[17];
    const float* bot_w   = (const float*)d_in[18];
    const float* bnb_g   = (const float*)d_in[20];
    const float* bnb_b   = (const float*)d_in[21];

    const int N = NN, H = HH;
    const int E = in_sizes[1] / 2;
    const int C = N / SUBG;
    const int* src = ei;
    const int* dst = ei + E;

    // workspace layout
    char* ws = (char*)d_ws;
    size_t off = 0;
    auto alloc = [&](size_t bytes) -> char* {
        char* p = ws + off;
        off = (off + bytes + 255) & ~(size_t)255;
        return p;
    };
    bf16_t* W1p  = (bf16_t*)alloc((size_t)H * H * 2);
    bf16_t* W2p  = (bf16_t*)alloc((size_t)H * H * 2);
    bf16_t* Wap  = (bf16_t*)alloc((size_t)H * (H / 2) * 2);
    bf16_t* Wfp  = (bf16_t*)alloc((size_t)H * H * 2);
    bf16_t* Wbp  = (bf16_t*)alloc((size_t)H * 32 * 2);
    float*  dis  = (float*)alloc((size_t)N * 4);
    bf16_t* Abf  = (bf16_t*)alloc((size_t)N * H * 2);   // bf16 activations (reused)
    float*  G    = (float*)alloc((size_t)N * H * 4);    // GEMM out (reused)
    float*  AGG  = (float*)alloc((size_t)N * H * 4);    // aggregation out (reused)
    float*  scr  = (float*)alloc((size_t)N * 4);        // attention scores
    bf16_t* Pbf  = (bf16_t*)alloc((size_t)C * H * 2);   // pooled (bf16)
    float*  Y1g  = (float*)alloc((size_t)C * H * 4);
    bf16_t* Y1bf = (bf16_t*)alloc((size_t)C * H * 2);
    float*  Y2g  = (float*)alloc((size_t)C * 32 * 4);
    float*  bns  = (float*)alloc(4 * 128 * 4);          // sum | sumsq | scale | shift
    float* bnsum = bns, *bnsq = bns + 128, *bnsc = bns + 256, *bnsh = bns + 384;

    float* out_y   = (float*)d_out;          // [C,32]
    float* out_emb = (float*)d_out + C * 32; // [N,H]

    // --- weight packing ---
    k_pack_w<<<CDIV(H * H, 256), 256, 0, stream>>>(conv1_w, W1p, H, H);
    k_pack_w<<<CDIV(H * H, 256), 256, 0, stream>>>(conv2_w, W2p, H, H);
    k_pack_w<<<CDIV(H * (H / 2), 256), 256, 0, stream>>>(att_w1, Wap, H, H / 2);
    k_pack_w<<<CDIV(H * H, 256), 256, 0, stream>>>(fc_w, Wfp, H, H);
    k_pack_w<<<CDIV(H * 32, 256), 256, 0, stream>>>(bot_w, Wbp, H, 32);

    // --- degree / deg^{-1/2} ---
    k_fill1<<<CDIV(N, 256), 256, 0, stream>>>(dis, N);
    k_deg_edges<<<CDIV(E, 256), 256, 0, stream>>>(dst, dis, E);
    k_rsqrt_inplace<<<CDIV(N, 256), 256, 0, stream>>>(dis, N);

    // --- x -> bf16 ---
    k_f32_to_bf16<<<CDIV(N * H, 256), 256, 0, stream>>>(x, Abf, N * H);

    const int strips_big = CDIV(CDIV(N, 16), 8);   // blocks for M=N strip GEMMs
    const int strips_C   = CDIV(CDIV(C, 16), 8);   // blocks for M=C strip GEMMs

    // ---- layer 1: GCNConv + BN + ReLU ----
    k_wmma_gemm_strip<8, 4><<<strips_big, 256, 0, stream>>>(Abf, W1p, G, N, dis, AGG);
    k_edge_agg<<<CDIV(E * H, 256), 256, 0, stream>>>(G, dis, src, dst, AGG, E * H);
    k_zero<<<1, 256, 0, stream>>>(bns, 256);
    k_bn_stats<<<CDIV(N, 256), H, 0, stream>>>(AGG, bnsum, bnsq, N, H, 256);
    k_bn_final<<<1, H, 0, stream>>>(bnsum, bnsq, bn1_g, bn1_b, bnsc, bnsh, N, H);
    k_bn_apply_relu<<<CDIV(N * H, 256), 256, 0, stream>>>(AGG, bnsc, bnsh,
                                                          nullptr, Abf, N, H);

    // ---- layer 2: GCNConv + BN + ReLU -> emb ----
    k_wmma_gemm_strip<8, 4><<<strips_big, 256, 0, stream>>>(Abf, W2p, G, N, dis, AGG);
    k_edge_agg<<<CDIV(E * H, 256), 256, 0, stream>>>(G, dis, src, dst, AGG, E * H);
    k_zero<<<1, 256, 0, stream>>>(bns, 256);
    k_bn_stats<<<CDIV(N, 256), H, 0, stream>>>(AGG, bnsum, bnsq, N, H, 256);
    k_bn_final<<<1, H, 0, stream>>>(bnsum, bnsq, bn2_g, bn2_b, bnsc, bnsh, N, H);
    k_bn_apply_relu<<<CDIV(N * H, 256), 256, 0, stream>>>(AGG, bnsc, bnsh,
                                                          out_emb, Abf, N, H);

    // ---- attention pooling ----
    k_wmma_gemm_strip<4, 4><<<strips_big, 256, 0, stream>>>(Abf, Wap, G, N,
                                                            nullptr, nullptr);
    k_scores<<<CDIV(N * 32, 256), 256, 0, stream>>>(G, att_b1, att_w2, scr, N);
    k_pool<<<C, 128, 0, stream>>>(out_emb, scr, Pbf);

    // ---- head: fc + BN + ReLU ----
    k_wmma_gemm_strip<8, 4><<<strips_C, 256, 0, stream>>>(Pbf, Wfp, Y1g, C,
                                                          nullptr, nullptr);
    k_zero<<<1, 256, 0, stream>>>(bns, 256);
    k_bn_stats<<<1, H, 0, stream>>>(Y1g, bnsum, bnsq, C, H, C);
    k_bn_final<<<1, H, 0, stream>>>(bnsum, bnsq, bne_g, bne_b, bnsc, bnsh, C, H);
    k_bn_apply_relu<<<CDIV(C * H, 256), 256, 0, stream>>>(Y1g, bnsc, bnsh,
                                                          nullptr, Y1bf, C, H);

    // ---- head: bottleneck + BN + ReLU -> y ----
    k_wmma_gemm_strip<2, 4><<<strips_C, 256, 0, stream>>>(Y1bf, Wbp, Y2g, C,
                                                          nullptr, nullptr);
    k_zero<<<1, 256, 0, stream>>>(bns, 256);
    k_bn_stats<<<1, 32, 0, stream>>>(Y2g, bnsum, bnsq, C, 32, C);
    k_bn_final<<<1, 32, 0, stream>>>(bnsum, bnsq, bnb_g, bnb_b, bnsc, bnsh, C, 32);
    k_bn_apply_relu<<<CDIV(C * 32, 256), 256, 0, stream>>>(Y2g, bnsc, bnsh,
                                                           out_y, nullptr, C, 32);

    (void)n_in; (void)out_size; (void)ws_size; (void)d_in;
}
